// PerceiverResampler_65996467470935
// MI455X (gfx1250) — matrix operand order repaired and checked by reading, not compiled, not run on hardware
//
#include <hip/hip_runtime.h>
#include <math.h>

// ---------------- problem constants ----------------
constexpr int cB = 32, cS = 576, cD = 1024, cL = 8, cH = 16, cHD = 64, cM = 4096, cLQ = 64;
#define EPSV 1e-6f

typedef __bf16 bf16;
typedef __attribute__((ext_vector_type(16))) __bf16 v16bf;
typedef __attribute__((ext_vector_type(4)))  __bf16 v4bf;
typedef __attribute__((ext_vector_type(8)))  float  v8f;

// gelu (tanh approx) via sigmoid identity: 0.5x(1+tanh(u)) = x * 1/(1+exp(-2u))
__device__ __forceinline__ float gelu_f(float x) {
  const float u2 = -1.5957691216057308f * (x + 0.044715f * x * x * x);
  return x * __builtin_amdgcn_rcpf(1.0f + __expf(u2));
}

// ---------------- WMMA fragment loaders (CDNA5 wave32 layouts) ----------------
// A matrix 16x32 bf16: lane&15 = row M, lane>>4 selects K-half.
// elements 0..7 -> K = 8*hi + j ; elements 8..15 -> K = 16 + 8*hi + j
__device__ __forceinline__ v16bf ld_fragA(const bf16* base, int stride, int lane) {
  const int r = lane & 15, hi = lane >> 4;
  const bf16* p = base + r * stride + hi * 8;
  v16bf f;
#pragma unroll
  for (int j = 0; j < 8; ++j) f[j] = p[j];
#pragma unroll
  for (int j = 0; j < 8; ++j) f[j + 8] = p[16 + j];
  return f;
}
// B matrix 32x16 bf16 staged transposed in LDS as [N][K]:
// lane&15 = column N, lane>>4 = K-half; elements 0..7 -> K = 8*hi + j ; elements 8..15 -> K = 16 + 8*hi + j
__device__ __forceinline__ v16bf ld_fragB(const bf16* base, int stride, int lane) {
  const int n = lane & 15, hi = lane >> 4;
  const bf16* p = base + n * stride + hi * 8;
  v16bf f;
#pragma unroll
  for (int j = 0; j < 8; ++j) f[j] = p[j];
#pragma unroll
  for (int j = 0; j < 8; ++j) f[j + 8] = p[16 + j];
  return f;
}

// ---------------- RMS norm (scale-only), float4 vectorized ----------------
__global__ __launch_bounds__(256)
void rmsnorm_kernel(const float* __restrict__ x, const float* __restrict__ scale,
                    float* __restrict__ out, int Dn) {
  __shared__ float red[256];
  const int row = blockIdx.x, tid = threadIdx.x;
  const float4* xr = (const float4*)(x + (size_t)row * Dn);
  const int n4 = Dn >> 2;
  float ss = 0.f;
  for (int i = tid; i < n4; i += 256) {
    const float4 v = xr[i];
    ss += v.x * v.x + v.y * v.y + v.z * v.z + v.w * v.w;
  }
  red[tid] = ss; __syncthreads();
  for (int s = 128; s > 0; s >>= 1) { if (tid < s) red[tid] += red[tid + s]; __syncthreads(); }
  const float inv = rsqrtf(red[0] / (float)Dn + EPSV);
  float4* o4 = (float4*)(out + (size_t)row * Dn);
  const float4* s4 = (const float4*)scale;
  for (int i = tid; i < n4; i += 256) {
    const float4 v = xr[i], sc = s4[i];
    float4 r;
    r.x = v.x * inv * sc.x; r.y = v.y * inv * sc.y;
    r.z = v.z * inv * sc.z; r.w = v.w * inv * sc.w;
    o4[i] = r;
  }
}

// ---------------- qk-norm over head_dim=64, one wave32 per row ----------------
__global__ __launch_bounds__(256)
void qknorm_kernel(float* __restrict__ x, const float* __restrict__ scale,
                   float post, int nrows) {
  const int lane = threadIdx.x & 31, wid = threadIdx.x >> 5;
  const int row = blockIdx.x * 8 + wid;
  if (row >= nrows) return;
  float2* xr = (float2*)(x + (size_t)row * 64);
  float2 v = xr[lane];
  float ss = v.x * v.x + v.y * v.y;
#pragma unroll
  for (int off = 16; off > 0; off >>= 1) ss += __shfl_xor(ss, off, 32);
  const float inv = rsqrtf(ss * (1.0f / 64.0f) + EPSV) * post;
  const float2 sc = ((const float2*)scale)[lane];
  v.x *= inv * sc.x; v.y *= inv * sc.y;
  xr[lane] = v;
}

// ---------------- latent broadcast (float4) ----------------
__global__ void broadcast_latents(const float4* __restrict__ lat_in, float4* __restrict__ lat_out) {
  const size_t i = (size_t)blockIdx.x * 256 + threadIdx.x;
  const size_t total4 = (size_t)cB * cLQ * cD / 4;
  if (i < total4) lat_out[i] = lat_in[i & ((size_t)cLQ * cD / 4 - 1)];  // LQ*D/4 = 2^14
}

// ---------------- tiled bf16 WMMA GEMM: C = epi(A @ B [+ R]) ----------------
// Block tile 128x128, K-step 32, 8 waves (4x2), 2 M-subtiles per wave.
// epi: 0=none, 1=+R, 2=tanh-GELU
#define GBM 128
#define GBN 128
#define GBK 32
__global__ __launch_bounds__(256)
void gemm_bf16_wmma(const float* __restrict__ A, const float* __restrict__ Bm,
                    const float* __restrict__ R, float* __restrict__ C,
                    int N, int K, int epi) {
  __shared__ __align__(16) bf16 As[GBM][GBK];
  __shared__ __align__(16) bf16 Bs[GBN][GBK];   // transposed: [N][K]
  const int tid = threadIdx.x, lane = tid & 31, wid = tid >> 5;
  const int wm = wid & 3, wn = wid >> 2;
  const int m0 = blockIdx.y * GBM, n0 = blockIdx.x * GBN;

  // staging maps (computed once; pointers advanced by constant strides)
  const int arow = tid >> 3, acol = (tid & 7) * 4;      // A: 32 rows/pass, 4 passes
  const int bkr  = tid >> 5, bnc  = (tid & 31) * 4;     // B: 8 k-rows/pass, 4 passes
  const float4* aptr = (const float4*)(A + (size_t)(m0 + arow) * K + acol);
  const float4* bptr = (const float4*)(Bm + (size_t)bkr * N + n0 + bnc);
  const size_t aPass = ((size_t)32 * K) >> 2;           // 32 rows, in float4 units
  const size_t bPass = ((size_t)8 * N) >> 2;            // 8 k-rows
  const size_t aStep = GBK >> 2;                        // +32 floats
  const size_t bStep = ((size_t)GBK * N) >> 2;          // +32 k-rows

  v8f acc0[4], acc1[4];
#pragma unroll
  for (int t = 0; t < 4; ++t)
#pragma unroll
    for (int e = 0; e < 8; ++e) { acc0[t][e] = 0.0f; acc1[t][e] = 0.0f; }

  for (int k0 = 0; k0 < K; k0 += GBK) {
    // phase 1: issue all global loads (8 x b128 outstanding)
    float4 areg[4], breg[4];
#pragma unroll
    for (int i = 0; i < 4; ++i) { areg[i] = aptr[i * aPass]; breg[i] = bptr[i * bPass]; }
    aptr += aStep; bptr += bStep;
    if (k0 + GBK < K) { __builtin_prefetch(aptr, 0, 1); __builtin_prefetch(bptr, 0, 1); }
    // phase 2: convert + store to LDS
#pragma unroll
    for (int i = 0; i < 4; ++i) {
      const v4bf ab = { (bf16)areg[i].x, (bf16)areg[i].y, (bf16)areg[i].z, (bf16)areg[i].w };
      *(v4bf*)&As[arow + 32 * i][acol] = ab;
      const int kr = bkr + 8 * i;
      Bs[bnc + 0][kr] = (bf16)breg[i].x;
      Bs[bnc + 1][kr] = (bf16)breg[i].y;
      Bs[bnc + 2][kr] = (bf16)breg[i].z;
      Bs[bnc + 3][kr] = (bf16)breg[i].w;
    }
    __syncthreads();

    const v16bf a0 = ld_fragA(&As[32 * wm][0], GBK, lane);
    const v16bf a1 = ld_fragA(&As[32 * wm + 16][0], GBK, lane);
#pragma unroll
    for (int t = 0; t < 4; ++t) {
      const v16bf b = ld_fragB(&Bs[64 * wn + 16 * t][0], GBK, lane);
      acc0[t] = __builtin_amdgcn_wmma_f32_16x16x32_bf16(false, a0, false, b,
                                                        (short)0, acc0[t], false, false);
      acc1[t] = __builtin_amdgcn_wmma_f32_16x16x32_bf16(false, a1, false, b,
                                                        (short)0, acc1[t], false, false);
    }
    __syncthreads();
  }

  const int hi = lane >> 4, lo = lane & 15;
  const int rowb = m0 + 32 * wm + 8 * hi;
  const int colb = n0 + 64 * wn + lo;
#pragma unroll
  for (int t = 0; t < 4; ++t) {
#pragma unroll
    for (int e = 0; e < 8; ++e) {
      const size_t idx0 = (size_t)(rowb + e) * N + colb + 16 * t;
      const size_t idx1 = idx0 + (size_t)16 * N;
      float v0 = acc0[t][e], v1 = acc1[t][e];
      if (epi == 1)      { v0 += R[idx0]; v1 += R[idx1]; }
      else if (epi == 2) { v0 = gelu_f(v0); v1 = gelu_f(v1); }
      C[idx0] = v0; C[idx1] = v1;
    }
  }
}

// ---------------- flash attention: block per (b,h), 64 queries, 64-key tiles ----------------
__global__ __launch_bounds__(256)
void attention_kernel(const float* __restrict__ qg, const float* __restrict__ kg,
                      const float* __restrict__ vg, const int* __restrict__ mask,
                      float* __restrict__ og, int Klen) {
  const int b = blockIdx.x / cH, h = blockIdx.x % cH;
  const int tid = threadIdx.x, lane = tid & 31, wid = tid >> 5;
  const int wm = wid & 3, wn = wid >> 2;          // waves: 4 row-groups x 2 col-groups
  const int hi = lane >> 4, lo = lane & 15;

  __shared__ __align__(16) bf16 Qs[64][64];       // A for S:  [q][d]
  __shared__ __align__(16) bf16 Ks[64][64];       // B for S:  [key][d]   ([N][K])
  __shared__ __align__(16) bf16 Vs[64][64];       // B for O:  [dim][key] ([N][K])
  __shared__ __align__(16) bf16 Ps[64][64];       // A for O:  [q][key]
  __shared__ float Ss[64][65];
  __shared__ float redm[64][4], reds[64][4];
  __shared__ float ms[64], ls[64], cs[64], kb[64];

  // per-thread staging map: 16 float4/row, 16 rows/pass, 4 passes
  const int d4 = (tid & 15) * 4, srow = tid >> 4;
  const size_t passStride = ((size_t)16 * cH * cHD) >> 2;   // 16 rows in float4 units
  const size_t tileStride = ((size_t)64 * cH * cHD) >> 2;   // 64 keys

  {  // stage Q once (already qk-normed, pre-scaled by HD^-0.5)
    const float4* qp = (const float4*)(qg + ((size_t)(b * cLQ + srow) * cH + h) * cHD + d4);
    float4 qv[4];
#pragma unroll
    for (int i = 0; i < 4; ++i) qv[i] = qp[i * passStride];
#pragma unroll
    for (int i = 0; i < 4; ++i) {
      const v4bf qb = { (bf16)qv[i].x, (bf16)qv[i].y, (bf16)qv[i].z, (bf16)qv[i].w };
      *(v4bf*)&Qs[srow + 16 * i][d4] = qb;
    }
  }
  if (tid < 64) { ms[tid] = -3.0e38f; ls[tid] = 0.0f; }

  const float4* kp = (const float4*)(kg + ((size_t)(b * Klen + srow) * cH + h) * cHD + d4);
  const float4* vp = (const float4*)(vg + ((size_t)(b * Klen + srow) * cH + h) * cHD + d4);
  const int* mp = (mask == nullptr) ? nullptr : mask + (size_t)b * Klen + (tid < 64 ? tid : 0);

  v8f acc[2];
#pragma unroll
  for (int t = 0; t < 2; ++t)
#pragma unroll
    for (int e = 0; e < 8; ++e) acc[t][e] = 0.0f;

  const int ntiles = Klen / 64;
  for (int kt = 0; kt < ntiles; ++kt) {
    // stage K-tile [key][d] and V-tile transposed [dim][key]; loads first, then cvt+store
    float4 kv[4], vv[4];
#pragma unroll
    for (int i = 0; i < 4; ++i) { kv[i] = kp[i * passStride]; vv[i] = vp[i * passStride]; }
    kp += tileStride; vp += tileStride;
#pragma unroll
    for (int i = 0; i < 4; ++i) {
      const int kr = srow + 16 * i;
      const v4bf kbv = { (bf16)kv[i].x, (bf16)kv[i].y, (bf16)kv[i].z, (bf16)kv[i].w };
      *(v4bf*)&Ks[kr][d4] = kbv;
      Vs[d4 + 0][kr] = (bf16)vv[i].x;
      Vs[d4 + 1][kr] = (bf16)vv[i].y;
      Vs[d4 + 2][kr] = (bf16)vv[i].z;
      Vs[d4 + 3][kr] = (bf16)vv[i].w;
    }
    if (tid < 64) kb[tid] = (mp == nullptr) ? 0.0f : (mp[kt * 64] > 0 ? 0.0f : -1e10f);
    __syncthreads();

    // S = Q @ K^T over d=64 (two chained 16x16x32 WMMAs per 16x16 tile)
    const v16bf qa0 = ld_fragA(&Qs[16 * wm][0], 64, lane);
    const v16bf qa1 = ld_fragA(&Qs[16 * wm][32], 64, lane);
#pragma unroll
    for (int t = 0; t < 2; ++t) {
      v8f s;
#pragma unroll
      for (int e = 0; e < 8; ++e) s[e] = 0.0f;
      const v16bf b0 = ld_fragB(&Ks[32 * wn + 16 * t][0], 64, lane);
      s = __builtin_amdgcn_wmma_f32_16x16x32_bf16(false, qa0, false, b0, (short)0, s, false, false);
      const v16bf b1 = ld_fragB(&Ks[32 * wn + 16 * t][32], 64, lane);
      s = __builtin_amdgcn_wmma_f32_16x16x32_bf16(false, qa1, false, b1, (short)0, s, false, false);
#pragma unroll
      for (int e = 0; e < 8; ++e) {
        const int rl = 16 * wm + e + 8 * hi, cl = 32 * wn + 16 * t + lo;
        Ss[rl][cl] = s[e] + kb[cl];
      }
    }
    __syncthreads();

    // online softmax, 4 threads per query row
    {
      const int r = tid >> 2, sg = tid & 3, c0 = sg * 16;
      float mx = -3.0e38f;
#pragma unroll
      for (int c = 0; c < 16; ++c) mx = fmaxf(mx, Ss[r][c0 + c]);
      redm[r][sg] = mx;
      __syncthreads();
      const float mo = ms[r];
      mx = fmaxf(fmaxf(redm[r][0], redm[r][1]), fmaxf(redm[r][2], redm[r][3]));
      mx = fmaxf(mx, mo);
      float sum = 0.0f;
#pragma unroll
      for (int c = 0; c < 16; ++c) {
        const float p = __expf(Ss[r][c0 + c] - mx);
        Ps[r][c0 + c] = (bf16)p;
        sum += p;
      }
      reds[r][sg] = sum;
      __syncthreads();
      if (sg == 0) {
        const float corr = __expf(mo - mx);
        ls[r] = ls[r] * corr + reds[r][0] + reds[r][1] + reds[r][2] + reds[r][3];
        ms[r] = mx;
        cs[r] = corr;
      }
    }
    __syncthreads();

    // rescale accumulators, then O += P @ V
#pragma unroll
    for (int t = 0; t < 2; ++t)
#pragma unroll
      for (int e = 0; e < 8; ++e) acc[t][e] *= cs[16 * wm + e + 8 * hi];

    const v16bf pa0 = ld_fragA(&Ps[16 * wm][0], 64, lane);
    const v16bf pa1 = ld_fragA(&Ps[16 * wm][32], 64, lane);
#pragma unroll
    for (int t = 0; t < 2; ++t) {
      const v16bf vb0 = ld_fragB(&Vs[32 * wn + 16 * t][0], 64, lane);
      acc[t] = __builtin_amdgcn_wmma_f32_16x16x32_bf16(false, pa0, false, vb0, (short)0, acc[t], false, false);
      const v16bf vb1 = ld_fragB(&Vs[32 * wn + 16 * t][32], 64, lane);
      acc[t] = __builtin_amdgcn_wmma_f32_16x16x32_bf16(false, pa1, false, vb1, (short)0, acc[t], false, false);
    }
    __syncthreads();
  }

  // write normalized output into [B, LQ, H*HD]
#pragma unroll
  for (int t = 0; t < 2; ++t)
#pragma unroll
    for (int e = 0; e < 8; ++e) {
      const int rl = 16 * wm + e + 8 * hi;
      const int dl = 32 * wn + 16 * t + lo;
      og[(size_t)(b * cLQ + rl) * (cH * cHD) + h * cHD + dl] = acc[t][e] / ls[rl];
    }
}

// ---------------- host orchestration ----------------
extern "C" void kernel_launch(void* const* d_in, const int* in_sizes, int n_in,
                              void* d_out, int out_size, void* d_ws, size_t ws_size,
                              hipStream_t stream) {
  (void)in_sizes; (void)n_in; (void)out_size; (void)ws_size;
  const float* embed      = (const float*)d_in[0];
  const int*   mask       = (const int*)  d_in[1];
  const float* latents    = (const float*)d_in[2];
  const float* ctx_scale  = (const float*)d_in[3];
  const float* perc_scale = (const float*)d_in[4];
  const float* pre_attn   = (const float*)d_in[5];
  const float* wq         = (const float*)d_in[6];
  const float* wk         = (const float*)d_in[7];
  const float* wv         = (const float*)d_in[8];
  const float* wo         = (const float*)d_in[9];
  const float* q_norm     = (const float*)d_in[10];
  const float* k_norm     = (const float*)d_in[11];
  const float* pre_mlp    = (const float*)d_in[12];
  const float* wi         = (const float*)d_in[13];
  const float* wo_mlp     = (const float*)d_in[14];

  const size_t SB = (size_t)cB * cS * cD;     // 18.9M floats
  const size_t LB = (size_t)cB * cLQ * cD;    // 2.1M floats
  float* ws   = (float*)d_ws;
  float* ctx  = ws;
  float* kbuf = ctx  + SB;
  float* vbuf = kbuf + SB;
  float* lat  = vbuf + SB;
  float* x_in = lat  + LB;
  float* qbuf = x_in + LB;
  float* attn = qbuf + LB;
  float* xbuf = attn + LB;
  float* ybuf = xbuf + LB;
  float* hbuf = ybuf + LB;                    // B*LQ*M floats

  const dim3 blk(256);

  rmsnorm_kernel<<<cB * cS, blk, 0, stream>>>(embed, ctx_scale, ctx, cD);
  broadcast_latents<<<(int)(LB / 4 / 256), blk, 0, stream>>>((const float4*)latents, (float4*)lat);

  for (int l = 0; l < cL; ++l) {
    const bool cross = (l % 2) == 0;          // XATTN = (0,2,4,6)
    const int  Klen  = cross ? cS : cLQ;
    const int  rows_kv = cB * Klen;
    const float* kvsrc = cross ? ctx : x_in;

    // pre-attention norm
    rmsnorm_kernel<<<cB * cLQ, blk, 0, stream>>>(lat, pre_attn + (size_t)l * cD, x_in, cD);

    // projections
    gemm_bf16_wmma<<<dim3(cD / GBN, (cB * cLQ) / GBM), blk, 0, stream>>>(
        x_in, wq + (size_t)l * cD * cD, nullptr, qbuf, cD, cD, 0);
    gemm_bf16_wmma<<<dim3(cD / GBN, rows_kv / GBM), blk, 0, stream>>>(
        kvsrc, wk + (size_t)l * cD * cD, nullptr, kbuf, cD, cD, 0);
    gemm_bf16_wmma<<<dim3(cD / GBN, rows_kv / GBM), blk, 0, stream>>>(
        kvsrc, wv + (size_t)l * cD * cD, nullptr, vbuf, cD, cD, 0);

    // qk-norm (+ fold HD^-0.5 into q)
    const int qrows = cB * cLQ * cH, krows = rows_kv * cH;
    qknorm_kernel<<<qrows / 8, blk, 0, stream>>>(qbuf, q_norm + (size_t)l * cHD, 0.125f, qrows);
    qknorm_kernel<<<krows / 8, blk, 0, stream>>>(kbuf, k_norm + (size_t)l * cHD, 1.0f, krows);

    // flash attention per (b,h)
    attention_kernel<<<cB * cH, blk, 0, stream>>>(qbuf, kbuf, vbuf,
                                                  cross ? mask : nullptr, attn, Klen);

    // output projection + residual
    gemm_bf16_wmma<<<dim3(cD / GBN, (cB * cLQ) / GBM), blk, 0, stream>>>(
        attn, wo + (size_t)l * cD * cD, lat, xbuf, cD, cD, 1);

    // MLP: norm -> GEMM+GELU -> GEMM+residual (writes new lat)
    rmsnorm_kernel<<<cB * cLQ, blk, 0, stream>>>(xbuf, pre_mlp + (size_t)l * cD, ybuf, cD);
    gemm_bf16_wmma<<<dim3(cM / GBN, (cB * cLQ) / GBM), blk, 0, stream>>>(
        ybuf, wi + (size_t)l * cD * cM, nullptr, hbuf, cM, cD, 2);
    gemm_bf16_wmma<<<dim3(cD / GBN, (cB * cLQ) / GBM), blk, 0, stream>>>(
        hbuf, wo_mlp + (size_t)l * cM * cD, xbuf, lat, cD, cM, 1);
  }

  rmsnorm_kernel<<<cB * cLQ, blk, 0, stream>>>(lat, perc_scale, (float*)d_out, cD);
}
